// GroupedQueryAttention_76948634075616
// MI455X (gfx1250) — matrix operand order repaired and checked by reading, not compiled, not run on hardware
//
#include <hip/hip_runtime.h>
#include <hip/hip_bf16.h>

// ---------------------------------------------------------------------------
// GroupedQueryAttention for MI455X (gfx1250, wave32, WMMA bf16, async-to-LDS)
//   B=2, S=2048, HIDDEN=2048, HEADS=32, KV_HEADS=8, D=64, GROUPS=4
// GEMMs + attention run through v_wmma_f32_16x16x32_bf16; flash-attention
// K/V tiles are double-buffered via global_load_async_to_lds_b128 (ASYNCcnt)
// so the DMA for tile i+1 runs under the WMMA/softmax work of tile i.
// ---------------------------------------------------------------------------

typedef __bf16 bf16_t;
typedef __attribute__((ext_vector_type(16))) __bf16 v16bf;
typedef __attribute__((ext_vector_type(8)))  __bf16 v8bf;
typedef __attribute__((ext_vector_type(4)))  __bf16 v4bf;
typedef __attribute__((ext_vector_type(8)))  float  v8f;

#define S_LEN   2048
#define HID     2048
#define KVD     512
#define DHEAD   64
#define NKVH    8
#define NGRP    4
#define BATCH   2

static __device__ __forceinline__ v8bf ld8(const bf16_t* p) {
    return *reinterpret_cast<const v8bf*>(p);
}
static __device__ __forceinline__ v16bf cat16(v8bf lo, v8bf hi) {
    v16bf r;
#pragma unroll
    for (int i = 0; i < 8; ++i) { r[i] = lo[i]; r[i + 8] = hi[i]; }
    return r;
}
static __device__ __forceinline__ v8f wmma_bf16(v16bf a, v16bf b, v8f c) {
    // D(f32 16x16) = A(bf16 16x32) * B(bf16 32x16) + C
    return __builtin_amdgcn_wmma_f32_16x16x32_bf16(
        false, a, false, b, (short)0, c, false, false);
}
// Raw 32-bit LDS byte offset (VDST operand of async-to-LDS ops).
static __device__ __forceinline__ unsigned lds_addr(const void* p) {
    return (unsigned)(size_t)(__attribute__((address_space(3))) const void*)p;
}
// Memory -> LDS async copy, 16 bytes per active lane (GVS addressing).
static __device__ __forceinline__ void async_load_b128(unsigned ldst,
                                                       unsigned goff,
                                                       unsigned long long sbase) {
    asm volatile("global_load_async_to_lds_b128 %0, %1, %2"
                 :: "v"(ldst), "v"(goff), "s"(sbase) : "memory");
}
static __device__ __forceinline__ void wait_asynccnt0() {
    asm volatile("s_wait_asynccnt 0" ::: "memory");
}

// ---------------------------------------------------------------------------
// GEMM: C[M,N] = A[M,K] * W[N,K]^T + bias[N]
//   A: fp32 (hidden) or bf16 (attention out). W/bias fp32.
//   CMODE 0: bf16 row-major [M,N]
//   CMODE 1: fp32 row-major [M,N]
//   CMODE 2: bf16 transposed per kv-head: vt[(b*NKVH+kvh)*64 + d][s]
// Block tile 32(M) x 256(N), K-chunk 64; 8 waves (2 M x 4 N), 16x64 per wave,
// 8 WMMAs per barrier pair.
// ---------------------------------------------------------------------------
template <bool A_F32, int CMODE>
__global__ __launch_bounds__(256)
void gqa_gemm_wmma(const void* __restrict__ Aptr,
                   const float* __restrict__ W,
                   const float* __restrict__ bias,
                   void* __restrict__ Cptr,
                   int M, int N, int K) {
    __shared__ bf16_t sA[32 * 72];    // 32 rows x 64 k (stride 72)
    __shared__ bf16_t sB[256 * 72];   // 256 n-rows x 64 k (stride 72)

    const int tid   = threadIdx.x;
    const int lane  = tid & 31;
    const int wave  = tid >> 5;
    const int waveM = wave >> 2;      // 0..1
    const int waveN = wave & 3;       // 0..3
    const int m0    = blockIdx.y * 32;
    const int n0    = blockIdx.x * 256;

    v8f acc[4];
#pragma unroll
    for (int j = 0; j < 4; ++j)
#pragma unroll
        for (int v = 0; v < 8; ++v) acc[j][v] = 0.0f;

    const float*  Af = (const float*)Aptr;
    const bf16_t* Ab = (const bf16_t*)Aptr;

    for (int kc = 0; kc < K; kc += 64) {
        // --- stage A tile (fp32->bf16 convert or bf16 copy): 32x64, 8/thread
        {
            const int r  = tid >> 3;          // 0..31
            const int c8 = (tid & 7) * 8;     // 0,8,..,56
#pragma unroll
            for (int half = 0; half < 2; ++half) {
                v4bf o;
                if (A_F32) {
                    const float4 f = *reinterpret_cast<const float4*>(
                        Af + (size_t)(m0 + r) * K + kc + c8 + half * 4);
                    o[0] = (bf16_t)f.x; o[1] = (bf16_t)f.y;
                    o[2] = (bf16_t)f.z; o[3] = (bf16_t)f.w;
                } else {
                    o = *reinterpret_cast<const v4bf*>(
                        Ab + (size_t)(m0 + r) * K + kc + c8 + half * 4);
                }
                *reinterpret_cast<v4bf*>(&sA[r * 72 + c8 + half * 4]) = o;
            }
        }
        // --- stage W tile (fp32->bf16): 256 rows x 64 k, one row per thread
        {
            const float* wrow = W + (size_t)(n0 + tid) * K + kc;
#pragma unroll
            for (int j = 0; j < 16; ++j) {
                const float4 f = *reinterpret_cast<const float4*>(wrow + j * 4);
                v4bf o;
                o[0] = (bf16_t)f.x; o[1] = (bf16_t)f.y;
                o[2] = (bf16_t)f.z; o[3] = (bf16_t)f.w;
                *reinterpret_cast<v4bf*>(&sB[tid * 72 + j * 4]) = o;
            }
        }
        __syncthreads();

        // --- A fragments for the two 32-wide k-chunks
        const int arow = waveM * 16 + (lane & 15);
        const int ko   = (lane >> 4) * 16;
        const bf16_t* ap = &sA[arow * 72];
        const v16bf a0 = cat16(ld8(ap + ko),      ld8(ap + ko + 8));
        const v16bf a1 = cat16(ld8(ap + 32 + ko), ld8(ap + 32 + ko + 8));

        // --- 8 B fragments + 8 WMMAs (B col n = W row n, K-major contiguous)
        const int co = (lane >> 4) * 16;
#pragma unroll
        for (int j = 0; j < 4; ++j) {
            const int brow = waveN * 64 + j * 16 + (lane & 15);
            const bf16_t* bp = &sB[brow * 72];
            const v16bf b0 = cat16(ld8(bp + co),      ld8(bp + co + 8));
            const v16bf b1 = cat16(ld8(bp + 32 + co), ld8(bp + 32 + co + 8));
            acc[j] = wmma_bf16(a0, b0, acc[j]);
            acc[j] = wmma_bf16(a1, b1, acc[j]);
        }
        __syncthreads();
    }

    // --- epilogue: bias + store (C layout: lane holds col=lane%16, rows v[+8])
#pragma unroll
    for (int j = 0; j < 4; ++j) {
        const int col = n0 + waveN * 64 + j * 16 + (lane & 15);
        const float bv = bias[col];
#pragma unroll
        for (int v = 0; v < 8; ++v) {
            const int row = m0 + waveM * 16 + (lane >> 4) * 8 + v;
            const float val = acc[j][v] + bv;
            if (CMODE == 1) {
                ((float*)Cptr)[(size_t)row * N + col] = val;
            } else if (CMODE == 0) {
                ((bf16_t*)Cptr)[(size_t)row * N + col] = (bf16_t)val;
            } else {
                // transposed V: vt[(b*NKVH + kvh)*64 + d][s], contiguous in s
                const int b_ = row >> 11;            // row / S_LEN
                const int s_ = row & (S_LEN - 1);
                const int kvh_ = col >> 6;
                const int d_   = col & 63;
                ((bf16_t*)Cptr)[(((size_t)(b_ * NKVH + kvh_) * DHEAD) + d_) * S_LEN + s_] =
                    (bf16_t)val;
            }
        }
    }
}

// ---------------------------------------------------------------------------
// Flash attention (GQA): block = one (b, kvh, g) head, 8 waves x 16 q-rows.
// Double-buffered async pipeline over 32 kv positions per step:
//   wait(tile i) -> barrier -> issue DMA(tile i+1) -> QK^T (4 WMMAs) ->
//   online softmax -> P relayout -> PV (4 WMMAs) -> barrier.
// ---------------------------------------------------------------------------
__global__ __launch_bounds__(256)
void gqa_flash_wmma(const bf16_t* __restrict__ qb,
                    const bf16_t* __restrict__ kb,
                    const bf16_t* __restrict__ vt,
                    const float*  __restrict__ mask,
                    bf16_t* __restrict__ ob) {
    __shared__ bf16_t sK[2][32 * 72];     // 32 kv rows x 64 dims (stride 72)
    __shared__ bf16_t sVt[2][64 * 40];    // V^T: 64 dims x 32 kv (stride 40)
    __shared__ bf16_t sP[8 * 16 * 40];    // per-wave P scratch 16x32 (stride 40)

    const int tid  = threadIdx.x;
    const int lane = tid & 31;
    const int wave = tid >> 5;

    const int b    = blockIdx.x >> 5;     // batch
    const int h    = blockIdx.x & 31;     // flat q-head = kvh*4 + g
    const int kvh  = h >> 2;
    const int q0   = blockIdx.y * 128 + wave * 16;

    // Block-uniform SGPR bases for async staging (GVS addressing)
    const unsigned long long kbase =
        (unsigned long long)(size_t)(kb + ((size_t)b * S_LEN) * KVD + kvh * DHEAD);
    const unsigned long long vtbase =
        (unsigned long long)(size_t)(vt + ((size_t)(b * NKVH + kvh) * DHEAD) * S_LEN);

    // Per-thread staging coordinates (fixed across iterations)
    const int kr = tid >> 3;              // kv row 0..31
    const int kd = (tid & 7) * 8;         // dim chunk
    const int vd = tid >> 2;              // dim 0..63
    const int vr = (tid & 3) * 8;         // kv chunk
    const unsigned kdst[2] = { lds_addr(&sK[0][kr * 72 + kd]),
                               lds_addr(&sK[1][kr * 72 + kd]) };
    const unsigned vdst[2] = { lds_addr(&sVt[0][vd * 40 + vr]),
                               lds_addr(&sVt[1][vd * 40 + vr]) };

    // Q fragments: 16 rows, D split into k-chunks [0..31], [32..63]
    const int ko = (lane >> 4) * 16;
    const size_t qbase = ((size_t)(b * S_LEN + q0 + (lane & 15))) * HID + h * DHEAD;
    const v16bf qa0 = cat16(ld8(qb + qbase + ko),      ld8(qb + qbase + ko + 8));
    const v16bf qa1 = cat16(ld8(qb + qbase + 32 + ko), ld8(qb + qbase + 32 + ko + 8));

    float mrow[8], lrow[8];
    v8f o[4];
#pragma unroll
    for (int v = 0; v < 8; ++v) { mrow[v] = -1e30f; lrow[v] = 0.0f; }
#pragma unroll
    for (int j = 0; j < 4; ++j)
#pragma unroll
        for (int v = 0; v < 8; ++v) o[j][v] = 0.0f;

    const float scale = 0.125f;           // 1/sqrt(64)
    bf16_t* pw = &sP[wave * (16 * 40)];
    const int co = (lane >> 4) * 16;
    const int NT = S_LEN / 32;

    // Prologue: start DMA for tile 0 into buffer 0
    async_load_b128(kdst[0], (unsigned)((kr * KVD + kd) * 2), kbase);
    async_load_b128(vdst[0], (unsigned)((vd * S_LEN + vr) * 2), vtbase);

    for (int it = 0; it < NT; ++it) {
        const int t0  = it * 32;
        const int buf = it & 1;

        // Tile `it`'s DMA was issued a full compute phase ago: finish + publish
        wait_asynccnt0();
        __syncthreads();

        // Kick off DMA for tile it+1 into the other buffer (runs under compute)
        if (it + 1 < NT) {
            const int nb = buf ^ 1;
            async_load_b128(kdst[nb], (unsigned)(((t0 + 32 + kr) * KVD + kd) * 2), kbase);
            async_load_b128(vdst[nb], (unsigned)((vd * S_LEN + t0 + 32 + vr) * 2), vtbase);
        }

        const bf16_t* cK  = sK[buf];
        const bf16_t* cVt = sVt[buf];

        // --- scores: frag0 = cols t0..t0+15, frag1 = cols t0+16..t0+31
        v8f s0, s1;
#pragma unroll
        for (int v = 0; v < 8; ++v) { s0[v] = 0.0f; s1[v] = 0.0f; }
        {
            const int kr0 = (lane & 15);
            const int kr1 = 16 + (lane & 15);
            const v16bf b00 = cat16(ld8(&cK[kr0 * 72 + co]),      ld8(&cK[kr0 * 72 + co + 8]));
            const v16bf b01 = cat16(ld8(&cK[kr0 * 72 + 32 + co]), ld8(&cK[kr0 * 72 + 32 + co + 8]));
            const v16bf b10 = cat16(ld8(&cK[kr1 * 72 + co]),      ld8(&cK[kr1 * 72 + co + 8]));
            const v16bf b11 = cat16(ld8(&cK[kr1 * 72 + 32 + co]), ld8(&cK[kr1 * 72 + 32 + co + 8]));
            s0 = wmma_bf16(qa0, b00, s0);
            s0 = wmma_bf16(qa1, b01, s0);
            s1 = wmma_bf16(qa0, b10, s1);
            s1 = wmma_bf16(qa1, b11, s1);
        }
        // --- scale + additive mask (mask indexed by key position)
        const float mk0 = mask[(size_t)b * S_LEN + t0 + (lane & 15)];
        const float mk1 = mask[(size_t)b * S_LEN + t0 + 16 + (lane & 15)];
#pragma unroll
        for (int v = 0; v < 8; ++v) {
            s0[v] = s0[v] * scale + mk0;
            s1[v] = s1[v] * scale + mk1;
        }
        // --- row max across the 16 lanes holding this row's columns
        float tmax[8];
#pragma unroll
        for (int v = 0; v < 8; ++v) tmax[v] = fmaxf(s0[v], s1[v]);
#pragma unroll
        for (int off = 1; off < 16; off <<= 1)
#pragma unroll
            for (int v = 0; v < 8; ++v)
                tmax[v] = fmaxf(tmax[v], __shfl_xor(tmax[v], off, 32));
        // --- online softmax update
        float corr[8], rsum[8];
#pragma unroll
        for (int v = 0; v < 8; ++v) {
            const float mn = fmaxf(mrow[v], tmax[v]);
            corr[v] = __expf(mrow[v] - mn);
            mrow[v] = mn;
            const float p0 = __expf(s0[v] - mn);
            const float p1 = __expf(s1[v] - mn);
            s0[v] = p0; s1[v] = p1;
            rsum[v] = p0 + p1;
        }
#pragma unroll
        for (int off = 1; off < 16; off <<= 1)
#pragma unroll
            for (int v = 0; v < 8; ++v)
                rsum[v] += __shfl_xor(rsum[v], off, 32);
#pragma unroll
        for (int v = 0; v < 8; ++v) lrow[v] = lrow[v] * corr[v] + rsum[v];
#pragma unroll
        for (int j = 0; j < 4; ++j)
#pragma unroll
            for (int v = 0; v < 8; ++v) o[j][v] *= corr[v];

        // --- P (C-layout) -> per-wave LDS -> reload in A-layout
        {
            const int pc = (lane & 15);
            const int rb = (lane >> 4) * 8;
#pragma unroll
            for (int v = 0; v < 8; ++v) {
                pw[(rb + v) * 40 + pc]      = (bf16_t)s0[v];
                pw[(rb + v) * 40 + pc + 16] = (bf16_t)s1[v];
            }
        }
        asm volatile("s_wait_dscnt 0" ::: "memory");  // per-wave LDS RAW fence
        const int prow = (lane & 15);
        const v16bf pa = cat16(ld8(pw + prow * 40 + ko), ld8(pw + prow * 40 + ko + 8));

        // --- O += P(16x32) * V(32x64): B col n = Vt row n (contiguous kv)
#pragma unroll
        for (int j = 0; j < 4; ++j) {
            const int vrow = j * 16 + (lane & 15);
            const v16bf bv = cat16(ld8(&cVt[vrow * 40 + co]), ld8(&cVt[vrow * 40 + co + 8]));
            o[j] = wmma_bf16(pa, bv, o[j]);
        }
        __syncthreads();
    }

    // --- normalize + store (attn output shares q's [b,s,h*64+d] layout)
    float inv[8];
#pragma unroll
    for (int v = 0; v < 8; ++v) inv[v] = 1.0f / lrow[v];
#pragma unroll
    for (int j = 0; j < 4; ++j) {
        const int col = j * 16 + (lane & 15);
#pragma unroll
        for (int v = 0; v < 8; ++v) {
            const int row = q0 + (lane >> 4) * 8 + v;
            ob[((size_t)(b * S_LEN + row)) * HID + h * DHEAD + col] =
                (bf16_t)(o[j][v] * inv[v]);
        }
    }
}

// ---------------------------------------------------------------------------
extern "C" void kernel_launch(void* const* d_in, const int* in_sizes, int n_in,
                              void* d_out, int out_size, void* d_ws, size_t ws_size,
                              hipStream_t stream) {
    const float* X    = (const float*)d_in[0];   // [2,2048,2048]
    const float* mask = (const float*)d_in[1];   // [2,2048]
    const float* Wq   = (const float*)d_in[2];
    const float* bq   = (const float*)d_in[3];
    const float* Wk   = (const float*)d_in[4];
    const float* bk   = (const float*)d_in[5];
    const float* Wv   = (const float*)d_in[6];
    const float* bv   = (const float*)d_in[7];
    const float* Wo   = (const float*)d_in[8];
    const float* bo   = (const float*)d_in[9];

    const int M = BATCH * S_LEN;  // 4096 rows

    char* ws = (char*)d_ws;
    bf16_t* qb = (bf16_t*)(ws);                                          // 16 MB
    bf16_t* kb = (bf16_t*)(ws + (size_t)M * HID * 2);                    //  4 MB
    bf16_t* vt = (bf16_t*)(ws + (size_t)M * HID * 2 + (size_t)M * KVD * 2);  // 4 MB (transposed)
    bf16_t* ob = (bf16_t*)(ws + (size_t)M * HID * 2 + 2 * (size_t)M * KVD * 2);

    dim3 blk(256);
    // Q/K projections (fp32 in, bf16 row-major out)
    gqa_gemm_wmma<true, 0><<<dim3(HID / 256, M / 32), blk, 0, stream>>>(
        X, Wq, bq, qb, M, HID, HID);
    gqa_gemm_wmma<true, 0><<<dim3(KVD / 256, M / 32), blk, 0, stream>>>(
        X, Wk, bk, kb, M, KVD, HID);
    // V projection writes kv-head-transposed layout for async V^T staging
    gqa_gemm_wmma<true, 2><<<dim3(KVD / 256, M / 32), blk, 0, stream>>>(
        X, Wv, bv, vt, M, KVD, HID);

    // Fused GQA flash attention: 64 heads x 16 q-row-tiles of 128
    gqa_flash_wmma<<<dim3(BATCH * 32, S_LEN / 128), blk, 0, stream>>>(
        qb, kb, vt, mask, ob);

    // Output projection (bf16 in, fp32 out)
    gqa_gemm_wmma<false, 1><<<dim3(HID / 256, M / 32), blk, 0, stream>>>(
        ob, Wo, bo, (float*)d_out, M, HID, HID);
}
